// FlashAttention3_41712722379427
// MI455X (gfx1250) — compile-verified, matrix operand order checked by reading, not run on hardware
//
#include <hip/hip_runtime.h>

// ---------------------------------------------------------------------------
// Causal multi-head self-attention (B=4, S=2048, D=1024, H=16, dh=64)
// bf16 WMMA (v_wmma_f32_16x16x32_bf16); f32 accumulate.
// Pipeline: cvt(x,W*) -> GEMM Q/K/V -> flash-attention -> GEMM out-proj.
// CDNA5 extras: global_load_async_to_lds_b128 staging (ASYNCcnt-tracked,
// double-buffered GEMM), global_prefetch_b8 hints.
// ---------------------------------------------------------------------------

#define DM    1024
#define HEADS 16
#define DH    64
#define SEQ   2048
#define BATCH 4
#define MTOT  (BATCH * SEQ)   // 8192 rows

typedef __attribute__((ext_vector_type(16))) __bf16       v16bf;
typedef __attribute__((ext_vector_type(8)))  float        v8f;
typedef __attribute__((ext_vector_type(4)))  unsigned int u32x4;
typedef __attribute__((ext_vector_type(4)))  int          v4i;

// ---- CDNA5 async global->LDS copy (ASYNCcnt) ------------------------------
#if defined(__has_builtin)
#if __has_builtin(__builtin_amdgcn_global_load_async_to_lds_b128)
#define HAVE_ASYNC_LDS 1
#endif
#endif

#ifdef HAVE_ASYNC_LDS
typedef __attribute__((address_space(1))) v4i as1_v4i;   // global int4
typedef __attribute__((address_space(3))) v4i as3_v4i;   // LDS int4
__device__ inline void async_cp_b128(const __bf16* g, __bf16* l) {
  __builtin_amdgcn_global_load_async_to_lds_b128(
      (as1_v4i*)g, (as3_v4i*)l, /*imm offset*/0, /*cpol*/0);
}
// copy 16 consecutive bf16 (32B) as two b128 async transfers
__device__ inline void async_cp16(const __bf16* g, __bf16* l) {
  async_cp_b128(g, l);
  async_cp_b128(g + 8, l + 8);
}
#if __has_builtin(__builtin_amdgcn_s_wait_asynccnt)
#define WAIT_ASYNC() __builtin_amdgcn_s_wait_asynccnt(0)
#else
#define WAIT_ASYNC() asm volatile("s_wait_asynccnt 0x0" ::: "memory")
#endif
#endif  // HAVE_ASYNC_LDS

union FragU { u32x4 q[2]; v16bf v; };

// Load a 16x32 bf16 WMMA operand fragment from row-major storage.
// A-operand: lane L<16 -> row L, K in {0..7,16..23}; lane>=16 -> row L-16,
// K in {8..15,24..31}.  The B operand (K x N, column n = row n of a row-major
// N x K matrix) uses the identical per-lane addressing.
__device__ inline v16bf load_frag(const __bf16* base, int stride) {
  const int lane = threadIdx.x & 31;
  const __bf16* p = base + (lane & 15) * stride + ((lane >> 4) << 3);
  FragU f;
  f.q[0] = *reinterpret_cast<const u32x4*>(p);
  f.q[1] = *reinterpret_cast<const u32x4*>(p + 16);
  return f.v;
}

__device__ inline v8f wmma_bf16(v16bf a, v16bf b, v8f c) {
  return __builtin_amdgcn_wmma_f32_16x16x32_bf16(
      /*neg_a=*/false, a, /*neg_b=*/false, b,
      /*c_mod=*/(short)0, c, /*reuse_a=*/false, /*reuse_b=*/false);
}

// ---------------------------------------------------------------------------
__global__ void cvt_bf16_kernel(const float* __restrict__ in,
                                __bf16* __restrict__ out, int n) {
  int i = blockIdx.x * blockDim.x + threadIdx.x;
  if (i < n) out[i] = (__bf16)in[i];
}

// ---------------------------------------------------------------------------
// C[m,n] = sum_k A[m,k] * W[n,k] + bias[n]   (torch Linear: A @ W^T + b)
// A: MxK bf16 row-major, W: NxK bf16 row-major.
// Block tile 128x128, 8 waves, each wave 32x64 (2x4 fragments), K step 32.
// Double-buffered LDS with async global->LDS copies when available.
__global__ __launch_bounds__(256) void gemm_bias(
    const __bf16* __restrict__ A, const __bf16* __restrict__ W,
    const float* __restrict__ bias, __bf16* __restrict__ outb,
    float* __restrict__ outf, int M, int N, int K) {
#ifdef HAVE_ASYNC_LDS
  __shared__ __bf16 As[2][128 * 32];
  __shared__ __bf16 Bs[2][128 * 32];
#else
  __shared__ __bf16 As[1][128 * 32];
  __shared__ __bf16 Bs[1][128 * 32];
#endif

  const int tid   = threadIdx.x;
  const int lane  = tid & 31;
  const int wave  = tid >> 5;
  const int wm    = wave & 3;             // 4 M-slots of 32 rows
  const int wn    = wave >> 2;            // 2 N-slots of 64 cols
  const int mbase = blockIdx.y * 128;
  const int nbase = blockIdx.x * 128;

  const v8f z8 = {0.f, 0.f, 0.f, 0.f, 0.f, 0.f, 0.f, 0.f};
  v8f acc[2][4];
#pragma unroll
  for (int i = 0; i < 2; ++i)
#pragma unroll
    for (int j = 0; j < 4; ++j) acc[i][j] = z8;

  const int srow = tid >> 1;              // staging row 0..127
  const int sch  = tid & 1;               // 16-element chunk within 32
  const __bf16* gA = A + (size_t)(mbase + srow) * K + sch * 16;
  const __bf16* gW = W + (size_t)(nbase + srow) * K + sch * 16;

#ifdef HAVE_ASYNC_LDS
  // prologue: stage k0=0 into buffer 0
  async_cp16(gA, As[0] + srow * 32 + sch * 16);
  async_cp16(gW, Bs[0] + srow * 32 + sch * 16);
  WAIT_ASYNC();
  __syncthreads();
  for (int k0 = 0; k0 < K; k0 += 32) {
    const int cur = (k0 >> 5) & 1;
    if (k0 + 32 < K) {  // issue next tile while computing this one
      async_cp16(gA + k0 + 32, As[cur ^ 1] + srow * 32 + sch * 16);
      async_cp16(gW + k0 + 32, Bs[cur ^ 1] + srow * 32 + sch * 16);
      if (k0 + 64 < K) {
        __builtin_prefetch(gA + k0 + 64, 0, 3);
        __builtin_prefetch(gW + k0 + 64, 0, 3);
      }
    }
    v16bf af[2];
#pragma unroll
    for (int mf = 0; mf < 2; ++mf)
      af[mf] = load_frag(As[cur] + (wm * 32 + mf * 16) * 32, 32);
#pragma unroll
    for (int nf = 0; nf < 4; ++nf) {
      v16bf bfrag = load_frag(Bs[cur] + (wn * 64 + nf * 16) * 32, 32);
#pragma unroll
      for (int mf = 0; mf < 2; ++mf)
        acc[mf][nf] = wmma_bf16(af[mf], bfrag, acc[mf][nf]);
    }
    WAIT_ASYNC();      // publish next tile's LDS writes
    __syncthreads();   // one barrier per K-step
  }
#else
  for (int k0 = 0; k0 < K; k0 += 32) {
    __syncthreads();
    {
      const u32x4* pa = reinterpret_cast<const u32x4*>(gA + k0);
      u32x4* da = reinterpret_cast<u32x4*>(As[0] + srow * 32 + sch * 16);
      da[0] = pa[0]; da[1] = pa[1];
      const u32x4* pb = reinterpret_cast<const u32x4*>(gW + k0);
      u32x4* db = reinterpret_cast<u32x4*>(Bs[0] + srow * 32 + sch * 16);
      db[0] = pb[0]; db[1] = pb[1];
      if (k0 + 32 < K) {
        __builtin_prefetch(gA + k0 + 32, 0, 3);
        __builtin_prefetch(gW + k0 + 32, 0, 3);
      }
    }
    __syncthreads();
    v16bf af[2];
#pragma unroll
    for (int mf = 0; mf < 2; ++mf)
      af[mf] = load_frag(As[0] + (wm * 32 + mf * 16) * 32, 32);
#pragma unroll
    for (int nf = 0; nf < 4; ++nf) {
      v16bf bfrag = load_frag(Bs[0] + (wn * 64 + nf * 16) * 32, 32);
#pragma unroll
      for (int mf = 0; mf < 2; ++mf)
        acc[mf][nf] = wmma_bf16(af[mf], bfrag, acc[mf][nf]);
    }
  }
#endif

  // epilogue: +bias, store bf16 or f32
  const int hl = lane >> 4;
  const int lc = lane & 15;
  const int row0 = mbase + wm * 32;
  const int col0 = nbase + wn * 64;
#pragma unroll
  for (int nf = 0; nf < 4; ++nf) {
    const int col = col0 + nf * 16 + lc;
    const float bv = bias[col];
#pragma unroll
    for (int mf = 0; mf < 2; ++mf) {
#pragma unroll
      for (int r = 0; r < 8; ++r) {
        const int row = row0 + mf * 16 + hl * 8 + r;
        const float val = acc[mf][nf][r] + bv;
        if (outb) outb[(size_t)row * N + col] = (__bf16)val;
        else      outf[(size_t)row * N + col] = val;
      }
    }
  }
}

// ---------------------------------------------------------------------------
// Flash attention, causal.  Q/K/V/O in flat [B*S, DM] bf16 layout (head h is
// the 64-column slice h*64..h*64+63).  One block = one (b,h) and a 128-row
// query tile; 8 waves x 16 query rows each.  KV tiles of 64 keys.
__global__ __launch_bounds__(256) void flash_attn(
    const __bf16* __restrict__ Qm, const __bf16* __restrict__ Km,
    const __bf16* __restrict__ Vm, __bf16* __restrict__ Om) {
  __shared__ __bf16 Kt[64 * 64];        // [key][d]
  __shared__ __bf16 Vt[64 * 64];        // [d][key]  (transposed for P@V)
  __shared__ __bf16 Ps[8 * 16 * 64];    // per-wave P scratch (C->A relayout)

  const int tid   = threadIdx.x;
  const int lane  = tid & 31;
  const int wave  = tid >> 5;
  const int bh    = blockIdx.y;
  const int b     = bh >> 4;
  const int h     = bh & 15;
  const int qb    = blockIdx.x * 128;   // query-tile start position
  const int qrow0 = qb + wave * 16;     // this wave's first query position

  // Q fragments resident for the whole kernel (d = 2 k-steps of 32)
  const size_t qoff = (size_t)(b * SEQ + qrow0) * DM + h * DH;
  v16bf aq[2];
#pragma unroll
  for (int ks = 0; ks < 2; ++ks) aq[ks] = load_frag(Qm + qoff + ks * 32, DM);

  const v8f z8 = {0.f, 0.f, 0.f, 0.f, 0.f, 0.f, 0.f, 0.f};
  v8f oacc[4];
#pragma unroll
  for (int d = 0; d < 4; ++d) oacc[d] = z8;
  float mrow[8], lrow[8];
#pragma unroll
  for (int r = 0; r < 8; ++r) { mrow[r] = -1.0e30f; lrow[r] = 0.0f; }

  const int hl = lane >> 4;
  const int lc = lane & 15;
  const int c16  = tid & 3;             // staging: which 16-col chunk
  const int srow = tid >> 2;            // staging: key row 0..63
  __bf16* pw = Ps + wave * (16 * 64);
  const int kvEnd = qb + 128;           // covers the diagonal for all rows

  for (int kv = 0; kv < kvEnd; kv += 64) {
    __syncthreads();
    {  // stage K tile (async if available) and transposed V tile
      const size_t g = (size_t)(b * SEQ + kv + srow) * DM + h * DH + c16 * 16;
#ifdef HAVE_ASYNC_LDS
      async_cp16(Km + g, Kt + srow * 64 + c16 * 16);
#else
      const u32x4* pk = reinterpret_cast<const u32x4*>(Km + g);
      u32x4* dk = reinterpret_cast<u32x4*>(Kt + srow * 64 + c16 * 16);
      dk[0] = pk[0]; dk[1] = pk[1];
#endif
      const u32x4* pv = reinterpret_cast<const u32x4*>(Vm + g);
      FragU fv; fv.q[0] = pv[0]; fv.q[1] = pv[1];
#pragma unroll
      for (int j = 0; j < 16; ++j)
        Vt[(c16 * 16 + j) * 64 + srow] = fv.v[j];
      if (kv + 64 < kvEnd) {
        __builtin_prefetch(Km + g + (size_t)64 * DM, 0, 3);
        __builtin_prefetch(Vm + g + (size_t)64 * DM, 0, 3);
      }
    }
#ifdef HAVE_ASYNC_LDS
    WAIT_ASYNC();
#endif
    __syncthreads();

    if (kv <= qrow0 + 15) {   // wave-uniform: tile not fully causally masked
      // ---- S = Q K^T : 16 x 64 scores ----
      v8f sc[4];
#pragma unroll
      for (int nf = 0; nf < 4; ++nf) {
        sc[nf] = z8;
#pragma unroll
        for (int ks = 0; ks < 2; ++ks) {
          v16bf bk = load_frag(Kt + (nf * 16) * 64 + ks * 32, 64);
          sc[nf] = wmma_bf16(aq[ks], bk, sc[nf]);
        }
      }
      // ---- scale (1/sqrt(64)) + causal mask ----
#pragma unroll
      for (int nf = 0; nf < 4; ++nf) {
        const int col = kv + nf * 16 + lc;
#pragma unroll
        for (int r = 0; r < 8; ++r) {
          const int row = qrow0 + hl * 8 + r;
          const float sv = sc[nf][r] * 0.125f;
          sc[nf][r] = (col > row) ? -3.0e38f : sv;
        }
      }
      // ---- row max (across 4 frags, then 16 lanes of each half) ----
      float mnew[8], alpha[8];
#pragma unroll
      for (int r = 0; r < 8; ++r) {
        float v = fmaxf(fmaxf(sc[0][r], sc[1][r]), fmaxf(sc[2][r], sc[3][r]));
#pragma unroll
        for (int m = 1; m < 16; m <<= 1) v = fmaxf(v, __shfl_xor(v, m, 32));
        mnew[r]  = fmaxf(mrow[r], v);
        alpha[r] = __expf(mrow[r] - mnew[r]);   // fully-masked tile: exp(0)=1
        mrow[r]  = mnew[r];
      }
#pragma unroll
      for (int d = 0; d < 4; ++d)
#pragma unroll
        for (int r = 0; r < 8; ++r) oacc[d][r] *= alpha[r];
      // ---- P = exp(S - m), spill to LDS in A-fragment friendly layout ----
      float rsum[8];
#pragma unroll
      for (int r = 0; r < 8; ++r) rsum[r] = 0.0f;
#pragma unroll
      for (int nf = 0; nf < 4; ++nf) {
#pragma unroll
        for (int r = 0; r < 8; ++r) {
          const float p = __expf(sc[nf][r] - mnew[r]);
          rsum[r] += p;
          pw[(hl * 8 + r) * 64 + nf * 16 + lc] = (__bf16)p;
        }
      }
#pragma unroll
      for (int r = 0; r < 8; ++r) {
        float v = rsum[r];
#pragma unroll
        for (int m = 1; m < 16; m <<= 1) v += __shfl_xor(v, m, 32);
        lrow[r] = lrow[r] * alpha[r] + v;
      }
      __builtin_amdgcn_wave_barrier();  // LDS is in-order within a wave
      // ---- O += P V ----
#pragma unroll
      for (int ks = 0; ks < 2; ++ks) {
        v16bf ap = load_frag(pw + ks * 32, 64);
#pragma unroll
        for (int d = 0; d < 4; ++d) {
          v16bf bv = load_frag(Vt + (d * 16) * 64 + ks * 32, 64);
          oacc[d] = wmma_bf16(ap, bv, oacc[d]);
        }
      }
    }
  }

  // ---- epilogue: O / l, store bf16 ----
#pragma unroll
  for (int d = 0; d < 4; ++d) {
    const int col = h * DH + d * 16 + lc;
#pragma unroll
    for (int r = 0; r < 8; ++r) {
      const int row = b * SEQ + qrow0 + hl * 8 + r;
      Om[(size_t)row * DM + col] = (__bf16)(oacc[d][r] / lrow[r]);
    }
  }
}

// ---------------------------------------------------------------------------
extern "C" void kernel_launch(void* const* d_in, const int* in_sizes, int n_in,
                              void* d_out, int out_size, void* d_ws, size_t ws_size,
                              hipStream_t stream) {
  const float* x  = (const float*)d_in[0];
  const float* Wq = (const float*)d_in[1];
  const float* bq = (const float*)d_in[2];
  const float* Wk = (const float*)d_in[3];
  const float* bk = (const float*)d_in[4];
  const float* Wv = (const float*)d_in[5];
  const float* bv = (const float*)d_in[6];
  const float* Wo = (const float*)d_in[7];
  const float* bo = (const float*)d_in[8];
  float* out = (float*)d_out;

  const size_t nX = (size_t)MTOT * DM;   // 8192*1024
  const size_t nW = (size_t)DM * DM;     // 1024*1024
  __bf16* xb = (__bf16*)d_ws;
  __bf16* wq = xb + nX;
  __bf16* wk = wq + nW;
  __bf16* wv = wk + nW;
  __bf16* wo = wv + nW;
  __bf16* Qb = wo + nW;
  __bf16* Kb = Qb + nX;
  __bf16* Vb = Kb + nX;
  __bf16* Ob = Vb + nX;

  const int T = 256;
  cvt_bf16_kernel<<<(unsigned)((nX + T - 1) / T), T, 0, stream>>>(x,  xb, (int)nX);
  cvt_bf16_kernel<<<(unsigned)((nW + T - 1) / T), T, 0, stream>>>(Wq, wq, (int)nW);
  cvt_bf16_kernel<<<(unsigned)((nW + T - 1) / T), T, 0, stream>>>(Wk, wk, (int)nW);
  cvt_bf16_kernel<<<(unsigned)((nW + T - 1) / T), T, 0, stream>>>(Wv, wv, (int)nW);
  cvt_bf16_kernel<<<(unsigned)((nW + T - 1) / T), T, 0, stream>>>(Wo, wo, (int)nW);

  dim3 gg(DM / 128, MTOT / 128);   // (8, 64)
  gemm_bias<<<gg, 256, 0, stream>>>(xb, wq, bq, Qb, nullptr, MTOT, DM, DM);
  gemm_bias<<<gg, 256, 0, stream>>>(xb, wk, bk, Kb, nullptr, MTOT, DM, DM);
  gemm_bias<<<gg, 256, 0, stream>>>(xb, wv, bv, Vb, nullptr, MTOT, DM, DM);

  flash_attn<<<dim3(SEQ / 128, BATCH * HEADS), 256, 0, stream>>>(Qb, Kb, Vb, Ob);

  gemm_bias<<<gg, 256, 0, stream>>>(Ob, wo, bo, nullptr, out, MTOT, DM, DM);
}
